// EmbeddingNet_83313775608156
// MI455X (gfx1250) — matrix operand order repaired and checked by reading, not compiled, not run on hardware
//
#include <hip/hip_runtime.h>

typedef __attribute__((ext_vector_type(16))) _Float16 v16h;
typedef __attribute__((ext_vector_type(8)))  _Float16 v8h;
typedef __attribute__((ext_vector_type(2)))  _Float16 v2h;
typedef __attribute__((ext_vector_type(8)))  float    v8f;
typedef __attribute__((ext_vector_type(4)))  float    v4f;

#define B_CONST   64
#define NMAX_     4096
#define D_        128    // NODE_DIM == 2*K
#define E_        256    // EMB
#define KHARM     64
#define ROWS_WG   128
#define WAVES_    8
#define TPB_      256
#define OUT_STRIDE 512   // 2*EMB
#define WELEMS    (D_ * E_)   // 32768 elements per weight matrix

// Map weight element (K=in-dim, E=out-dim) to WMMA B-operand swizzled index:
// lane l<16 holds col E%16, K-chunk elems 0..15; lane l>=16 holds K elems 16..31.
__device__ __forceinline__ int swizzleB(int Kk, int Ee) {
    const int nt = Ee >> 4, l0 = Ee & 15;
    const int kc = Kk >> 5, kin = Kk & 31;
    const int ln = (kin < 16) ? l0 : (l0 + 16);
    return (((nt << 2) + kc) * 32 + ln) * 16 + (kin & 15);
}

// One-time weight prep: f32 -> f16, pre-swizzled into B-operand layout, in d_ws.
// ws[0..32767] = W_node, ws[32768..65535] = W_cyc  (128 KB total)
__global__ __launch_bounds__(TPB_)
void prep_weights(const float* __restrict__ Wn, const float* __restrict__ Wc,
                  _Float16* __restrict__ ws) {
    const int idx = blockIdx.x * TPB_ + threadIdx.x;   // [0, 65536)
    const int mat = idx >> 15;
    const int i   = idx & (WELEMS - 1);
    const float w = mat ? Wc[i] : Wn[i];
    ws[mat * WELEMS + swizzleB(i >> 8, i & 255)] = (_Float16)w;
}

__global__ __launch_bounds__(TPB_)
void embednet_fused(const float* __restrict__ x,
                    const int*   __restrict__ lengths,
                    const _Float16* __restrict__ wsW,
                    float* __restrict__ out)
{
    // Both weight matrices, already in B-operand layout (f16): 128 KB
    __shared__ __align__(16) _Float16 sW2[2 * WELEMS];
    // Per-wave activation tile (x_norm, then ce_raw), row-major [16][128] f16: 32 KB
    __shared__ __align__(16) _Float16 sA[WAVES_][16][D_];

    const int tid  = threadIdx.x;
    const int wave = tid >> 5;
    const int lane = tid & 31;
    const int blk  = blockIdx.x;
    const int b    = blk >> 5;               // 32 row-blocks per batch
    const int n0   = (blk & 31) * ROWS_WG;

    // ---- Phase 0a: kick off async 128 KB weight copy ws -> LDS (ASYNCcnt) ----
    {
        const unsigned ldsBase = (unsigned)(uintptr_t)&sW2[0];   // low 32 bits = LDS offset
        const unsigned long long gBase = (unsigned long long)(uintptr_t)wsW;
        #pragma unroll 8
        for (int it = 0; it < 32; ++it) {
            const unsigned off = (unsigned)(tid * 16 + it * 4096);  // 16B per thread per iter
            const unsigned laddr = ldsBase + off;
            const unsigned long long gaddr = gBase + off;
            asm volatile("global_load_async_to_lds_b128 %0, %1, off"
                         :: "v"(laddr), "v"(gaddr) : "memory");
        }
    }

    const int   len  = lengths[b];
    const float Nb   = (float)(len > 1 ? len : 1);
    int keff = len >> 1;
    keff = keff < 1 ? 1 : (keff > KHARM ? KHARM : keff);
    const float kscale = sqrtf((float)KHARM / (float)keff);

    // ---- Phase 0b: LayerNorm of 16 rows per wave -> sA (overlaps async copy) ----
    {
        const int m    = lane & 15;        // row within wave tile
        const int half = lane >> 4;        // 2 lanes per row, 64 floats each
        const int row  = n0 + wave * 16 + m;
        const float* xr = x + ((size_t)b * NMAX_ + row) * D_ + half * 64;

        float s = 0.f, s2 = 0.f;
        #pragma unroll
        for (int i = 0; i < 16; ++i) {
            v4f t = *(const v4f*)(xr + i * 4);
            #pragma unroll
            for (int j = 0; j < 4; ++j) { s += t[j]; s2 += t[j] * t[j]; }
        }
        s  += __shfl_xor(s, 16, 32);
        s2 += __shfl_xor(s2, 16, 32);
        const float mean = s * (1.0f / 128.0f);
        const float var  = s2 * (1.0f / 128.0f) - mean * mean;
        const float rstd = rsqrtf(var + 1e-5f);

        #pragma unroll
        for (int i = 0; i < 8; ++i) {    // 8 chunks of 8 floats -> one 16B LDS store
            v4f t0 = *(const v4f*)(xr + i * 8);
            v4f t1 = *(const v4f*)(xr + i * 8 + 4);
            v8h hv;
            #pragma unroll
            for (int j = 0; j < 4; ++j) {
                hv[j]     = (_Float16)((t0[j] - mean) * rstd);
                hv[j + 4] = (_Float16)((t1[j] - mean) * rstd);
            }
            *(v8h*)(&sA[wave][m][half * 64 + i * 8]) = hv;
        }
    }

    // Weights landed; make visible to all waves. Only barrier in the kernel.
    asm volatile("s_wait_asynccnt 0x0" ::: "memory");
    __syncthreads();

    // One 16x(K=128)x256 GEMM from sA (own wave tile) and an LDS weight buffer.
    auto run_gemm = [&](const _Float16* wbase, int colOff) {
        // A operand, contiguous K order (matches B swizzle):
        // lanes 0-15: M=lane, K = {0..15} per 32-chunk
        // lanes 16-31: M=lane-16, K = {16..31} per 32-chunk
        const int m  = lane & 15;
        const int kb = (lane < 16) ? 0 : 16;
        const _Float16* ar = &sA[wave][m][0];
        v16h a[4];
        #pragma unroll
        for (int kc = 0; kc < 4; ++kc) {
            v8h lo = *(const v8h*)(ar + kc * 32 + kb);
            v8h hi = *(const v8h*)(ar + kc * 32 + kb + 8);
            #pragma unroll
            for (int j = 0; j < 8; ++j) { a[kc][j] = lo[j]; a[kc][j + 8] = hi[j]; }
        }

        const int cl = lane & 15;
        #pragma unroll 4
        for (int nt = 0; nt < 16; ++nt) {
            v8f c = {};
            #pragma unroll
            for (int kc = 0; kc < 4; ++kc) {
                const _Float16* bp = wbase + (((nt << 2) + kc) * 32 + lane) * 16;
                v8h blo = *(const v8h*)bp;
                v8h bhi = *(const v8h*)(bp + 8);
                v16h bb;
                #pragma unroll
                for (int j = 0; j < 8; ++j) { bb[j] = blo[j]; bb[j + 8] = bhi[j]; }
                c = __builtin_amdgcn_wmma_f32_16x16x32_f16(
                        false, a[kc], false, bb, (short)0, c, false, false);
            }
            const int col = colOff + nt * 16 + cl;
            #pragma unroll
            for (int j = 0; j < 8; ++j) {
                const int row = n0 + wave * 16 + j + ((lane < 16) ? 0 : 8);
                out[((size_t)b * NMAX_ + row) * OUT_STRIDE + col] = c[j];
            }
        }
    };

    // ---- Phase 1: nfe = LN(x) @ W_node -> cols [0,256) ----
    run_gemm(&sW2[0], 0);

    // ---- Phase 2: ce_raw -> own wave's sA (intra-wave dep only, no barrier) ----
    {
        const float twoPiOverN = 6.28318530717958647692f / Nb;
        for (int idx = lane; idx < 16 * KHARM; idx += 32) {
            const int m = idx >> 6;
            const int j = idx & 63;
            const int h = j + 1;
            const int n = n0 + wave * 16 + m;
            float coef = 0.f;
            if (n < len && h <= keff) coef = rsqrtf((float)h) * kscale;
            const float ang = twoPiOverN * (float)n * (float)h;
            v2h p;
            p[0] = (_Float16)(__cosf(ang) * coef);
            p[1] = (_Float16)(__sinf(ang) * coef);
            *(v2h*)(&sA[wave][m][2 * j]) = p;   // interleaved (cos_h, sin_h)
        }
    }

    // ---- Phase 3: ce = ce_raw @ W_cyc -> cols [256,512) ----
    run_gemm(&sW2[WELEMS], E_);

    // ---- t_mask output (float 0/1), after feats block ----
    if (tid < ROWS_WG) {
        float* mask = out + (size_t)B_CONST * NMAX_ * OUT_STRIDE;
        mask[(size_t)b * NMAX_ + n0 + tid] = ((n0 + tid) < len) ? 1.0f : 0.0f;
    }
}

extern "C" void kernel_launch(void* const* d_in, const int* in_sizes, int n_in,
                              void* d_out, int out_size, void* d_ws, size_t ws_size,
                              hipStream_t stream) {
    (void)in_sizes; (void)n_in; (void)out_size; (void)ws_size;  // needs >= 128 KB scratch
    const float* x       = (const float*)d_in[0];
    const int*   lengths = (const int*)d_in[1];
    const float* Wn      = (const float*)d_in[2];
    const float* Wc      = (const float*)d_in[3];
    _Float16*    ws      = (_Float16*)d_ws;
    float*       out     = (float*)d_out;

    // One-time (per launch) weight convert+swizzle into scratch: 65536 elements.
    prep_weights<<<dim3(WELEMS * 2 / TPB_), dim3(TPB_), 0, stream>>>(Wn, Wc, ws);

    // Main fused kernel: 64 batches x 32 row-blocks = 2048 workgroups.
    embednet_fused<<<dim3(B_CONST * (NMAX_ / ROWS_WG)), dim3(TPB_), 0, stream>>>(
        x, lengths, ws, out);
}
